// CondGenerator_14963666059844
// MI455X (gfx1250) — matrix operand order repaired and checked by reading, not compiled, not run on hardware
//
#include <hip/hip_runtime.h>
#include <cmath>

// Problem constants (from reference)
#define BB 16
#define SSQ 512
#define EE 512
#define HH 8
#define LLAY 6
#define FF 2048
#define HDIM 64
#define VV 100
#define NGEN 80
#define NRING 20
#define MROWS (BB * SSQ)          // 8192
#define NEGV -1000000000.0f
#define PAD_ID 99
#define RING_ID 1

typedef __attribute__((ext_vector_type(16))) __bf16 v16bf;
typedef __attribute__((ext_vector_type(8)))  float  v8f;

__device__ __forceinline__ __bf16 f2bf(float f) {
  unsigned u = __builtin_bit_cast(unsigned, f);
  unsigned r = (u + 0x7fffu + ((u >> 16) & 1u)) >> 16;
  unsigned short s = (unsigned short)r;
  return __builtin_bit_cast(__bf16, s);
}

__device__ __forceinline__ v8f wmma_bf16(v16bf a, v16bf b, v8f c) {
  // v_wmma_f32_16x16x32_bf16: D = A(16x32) * B(32x16) + C
  return __builtin_amdgcn_wmma_f32_16x16x32_bf16(false, a, false, b, (short)0, c, false, false);
}

// ---------------------------------------------------------------------------
// fp32 -> bf16 conversion (weights)
// ---------------------------------------------------------------------------
__global__ void __launch_bounds__(256) cvt_kernel(const float* __restrict__ in,
                                                  __bf16* __restrict__ out, long n) {
  long t = (long)blockIdx.x * blockDim.x + threadIdx.x;
  if (t < n) out[t] = f2bf(in[t]);
}

// ---------------------------------------------------------------------------
// Embedding: x = sqrtE*(tok_emb[seq] + count_emb[cnt]); write f32 + bf16
// ---------------------------------------------------------------------------
__global__ void __launch_bounds__(256) embed_kernel(
    const int* __restrict__ seq, const int* __restrict__ cnt,
    const float* __restrict__ tok_emb, const float* __restrict__ count_emb,
    float* __restrict__ xf, __bf16* __restrict__ xbf) {
  long t = (long)blockIdx.x * blockDim.x + threadIdx.x;
  if (t >= (long)MROWS * EE) return;
  int e = (int)(t % EE);
  long bs = t / EE;
  int tok = seq[bs];
  int c = cnt[bs];
  float v = (tok_emb[(size_t)tok * EE + e] + count_emb[(size_t)c * EE + e]) * 22.62741699796952f;
  xf[t] = v;
  xbf[t] = f2bf(v);
}

// ---------------------------------------------------------------------------
// Attention additive mask: (B,H,S,S) = causal? (lin+up+down bias) : NEG, +pad
// ---------------------------------------------------------------------------
__global__ void __launch_bounds__(256) build_mask_kernel(
    const int* __restrict__ lin, const int* __restrict__ up, const int* __restrict__ down,
    const float* __restrict__ lemb, const float* __restrict__ uemb, const float* __restrict__ demb,
    const int* __restrict__ seq, float* __restrict__ mask) {
  long t = (long)blockIdx.x * blockDim.x + threadIdx.x;
  if (t >= (long)BB * SSQ * SSQ) return;
  int k = (int)(t % SSQ);
  long r = t / SSQ;
  int q = (int)(r % SSQ);
  int b = (int)(r / SSQ);
  int li = lin[t], ui = up[t], di = down[t];
  bool causal = (k <= q);
  float padadd = (seq[(size_t)b * SSQ + k] == PAD_ID) ? NEGV : 0.0f;
#pragma unroll
  for (int h = 0; h < HH; ++h) {
    float bias = lemb[(size_t)li * HH + h] + uemb[(size_t)ui * HH + h] + demb[(size_t)di * HH + h];
    float val = (causal ? bias : NEGV) + padadd;
    mask[((((size_t)b * HH + h) * SSQ + q) * SSQ) + k] = val;
  }
}

// ---------------------------------------------------------------------------
// V repack: qkv(B*S,3E) -> vt(B,H,HD,S) bf16 so P@V B-fragments are contiguous
// ---------------------------------------------------------------------------
__global__ void __launch_bounds__(256) repack_v_kernel(
    const __bf16* __restrict__ qkv, __bf16* __restrict__ vt) {
  long t = (long)blockIdx.x * blockDim.x + threadIdx.x;
  if (t >= (long)BB * HH * HDIM * SSQ) return;
  int s = (int)(t % SSQ);
  long r = t / SSQ;
  int d = (int)(r % HDIM);
  r /= HDIM;
  int h = (int)(r % HH);
  int b = (int)(r / HH);
  vt[t] = qkv[((size_t)b * SSQ + s) * (3 * EE) + 2 * EE + h * HDIM + d];
}

// ---------------------------------------------------------------------------
// WMMA GEMM: out[M,N] = A[M,K](bf16) @ W[N,K]^T(bf16) + bias, optional GELU.
// Block: 128 threads = 4 waves; block tile 64 rows x 64 cols; wave = 16x64.
// B tile (64 cols x 32 K) staged to LDS with global_load_async_to_lds_b128
// (ASYNCcnt), shared by all 4 waves.
// ---------------------------------------------------------------------------
template <bool FULL>
__global__ void __launch_bounds__(128) wmma_gemm_kernel(
    const __bf16* __restrict__ A, const __bf16* __restrict__ W,
    const float* __restrict__ bias, float* __restrict__ outF,
    __bf16* __restrict__ outB, int M, int N, int K, int ldc, int act) {
  __shared__ __bf16 Bsh[64][32];   // [col][k]
  int tid = threadIdx.x;
  int lane = tid & 31;
  int wave = tid >> 5;
  int hi = lane >> 4;
  int ln15 = lane & 15;
  int nBase = blockIdx.x * 64;
  int mBase = blockIdx.y * 64 + wave * 16;

  // staging: 64 cols x 64 bytes = 256 16B chunks; 2 chunks per thread
  int c0 = tid >> 2, q0 = tid & 3;
  int c1 = (tid + 128) >> 2, q1 = tid & 3;
  int n0 = nBase + c0, n1 = nBase + c1;
  if (!FULL) {
    n0 = min(n0, N - 1);
    n1 = min(n1, N - 1);
  }
  unsigned l0 = (unsigned)(uintptr_t)&Bsh[c0][q0 * 8];
  unsigned l1 = (unsigned)(uintptr_t)&Bsh[c1][q1 * 8];
  const __bf16* g0base = W + (size_t)n0 * K + q0 * 8;
  const __bf16* g1base = W + (size_t)n1 * K + q1 * 8;

  v8f zero = {0.f, 0.f, 0.f, 0.f, 0.f, 0.f, 0.f, 0.f};
  v8f acc[4] = {zero, zero, zero, zero};

  const __bf16* Arow = A + (size_t)(mBase + ln15) * K;
  for (int k0 = 0; k0 < K; k0 += 32) {
    // async-stage B tile into LDS (per-lane 16B copies, ASYNCcnt tracked)
    {
      unsigned long long ga = (unsigned long long)(g0base + k0);
      unsigned long long gb = (unsigned long long)(g1base + k0);
      asm volatile("global_load_async_to_lds_b128 %0, %1, off"
                   :: "v"(l0), "v"(ga) : "memory");
      asm volatile("global_load_async_to_lds_b128 %0, %1, off"
                   :: "v"(l1), "v"(gb) : "memory");
    }

    // A fragment from global while the async copy is in flight
    v16bf a;
    const __bf16* Ap = Arow + k0;
#pragma unroll
    for (int j = 0; j < 8; ++j) a[j] = Ap[hi * 16 + j];
#pragma unroll
    for (int j = 0; j < 8; ++j) a[8 + j] = Ap[hi * 16 + 8 + j];
    if (k0 + 32 < K) __builtin_prefetch(Ap + 32, 0, 3);

    asm volatile("s_wait_asynccnt 0x0" ::: "memory");
    __syncthreads();

#pragma unroll
    for (int c = 0; c < 4; ++c) {
      const __bf16* Wp = &Bsh[c * 16 + ln15][hi * 16];
      v16bf bfr;
#pragma unroll
      for (int j = 0; j < 16; ++j) bfr[j] = Wp[j];
      acc[c] = wmma_bf16(a, bfr, acc[c]);
    }
    __syncthreads();
  }

#pragma unroll
  for (int c = 0; c < 4; ++c) {
    int n = nBase + c * 16 + ln15;
    if (!FULL && n >= N) continue;
    float bv = bias ? bias[n] : 0.0f;
#pragma unroll
    for (int i = 0; i < 8; ++i) {
      int m = mBase + hi * 8 + i;
      float v = acc[c][i] + bv;
      if (act == 1) v = 0.5f * v * (1.0f + erff(v * 0.70710678118654752f));
      if (outF) outF[(size_t)m * ldc + n] = v;
      if (outB) outB[(size_t)m * ldc + n] = f2bf(v);
    }
  }
}

// ---------------------------------------------------------------------------
// Fused flash attention. One wave per (b, h, 16-query tile). 4 waves/block.
// qkv: (B*S, 3E) bf16.  vt: (B,H,HD,S) bf16.  mask: (B,H,S,S) f32.
// out: (B*S, E) bf16.
// ---------------------------------------------------------------------------
__global__ void __launch_bounds__(128) attn_kernel(
    const __bf16* __restrict__ qkv, const __bf16* __restrict__ vt,
    const float* __restrict__ mask, __bf16* __restrict__ attnOut) {
  __shared__ __bf16 pbuf[4][16][32];
  int lane = threadIdx.x & 31;
  int wave = threadIdx.x >> 5;
  int hi = lane >> 4;
  int ln15 = lane & 15;
  int wid = blockIdx.x * 4 + wave;
  int qt = wid & 31;                 // S/16 = 32 tiles
  int h = (wid >> 5) & 7;
  int b = wid >> 8;
  int qBase = qt * 16;
  size_t bsBase = (size_t)b * SSQ;
  const int ld = 3 * EE;

  // Q fragments (16 queries x 64 head-dim -> two 16x32 A fragments)
  v16bf aQ[2];
  {
    const __bf16* Qr = qkv + (bsBase + qBase + ln15) * ld + h * HDIM;
#pragma unroll
    for (int t = 0; t < 2; ++t) {
      int k0 = t * 32;
#pragma unroll
      for (int j = 0; j < 8; ++j) aQ[t][j] = Qr[k0 + hi * 16 + j];
#pragma unroll
      for (int j = 0; j < 8; ++j) aQ[t][8 + j] = Qr[k0 + hi * 16 + 8 + j];
    }
  }

  float rowMax[8], rowSum[8];
#pragma unroll
  for (int i = 0; i < 8; ++i) { rowMax[i] = -3.0e38f; rowSum[i] = 0.0f; }
  v8f zero = {0.f, 0.f, 0.f, 0.f, 0.f, 0.f, 0.f, 0.f};
  v8f accO[4] = {zero, zero, zero, zero};

  const float scale = 0.125f; // 1/sqrt(64)
  const float* mrow = mask + (((size_t)b * HH + h) * SSQ + qBase) * SSQ;
  const __bf16* vbase = vt + ((size_t)b * HH + h) * HDIM * SSQ;

  for (int kt = 0; kt < SSQ; kt += 32) {
    v8f s0 = zero, s1 = zero;
    // scores: Q(16x64) @ K^T(64x32) via 2 K-steps x 2 key tiles
#pragma unroll
    for (int t = 0; t < 2; ++t) {
      int kd0 = t * 32;
      v16bf b0, b1;
      const __bf16* Kr0 = qkv + (bsBase + kt + ln15) * ld + EE + h * HDIM + kd0 + hi * 16;
      const __bf16* Kr1 = qkv + (bsBase + kt + 16 + ln15) * ld + EE + h * HDIM + kd0 + hi * 16;
#pragma unroll
      for (int j = 0; j < 16; ++j) b0[j] = Kr0[j];
#pragma unroll
      for (int j = 0; j < 16; ++j) b1[j] = Kr1[j];
      s0 = wmma_bf16(aQ[t], b0, s0);
      s1 = wmma_bf16(aQ[t], b1, s1);
    }
    // online softmax per query row (rows live on 16-lane halves)
    float rscale[8];
#pragma unroll
    for (int i = 0; i < 8; ++i) {
      int m = hi * 8 + i;
      float v0 = s0[i] * scale + mrow[(size_t)m * SSQ + kt + ln15];
      float v1 = s1[i] * scale + mrow[(size_t)m * SSQ + kt + 16 + ln15];
      float tmax = fmaxf(v0, v1);
#pragma unroll
      for (int off = 1; off < 16; off <<= 1) tmax = fmaxf(tmax, __shfl_xor(tmax, off, 32));
      float nm = fmaxf(rowMax[i], tmax);
      float r = __expf(rowMax[i] - nm);
      float p0 = __expf(v0 - nm);
      float p1 = __expf(v1 - nm);
      float ps = p0 + p1;
#pragma unroll
      for (int off = 1; off < 16; off <<= 1) ps += __shfl_xor(ps, off, 32);
      rowSum[i] = rowSum[i] * r + ps;
      rowMax[i] = nm;
      rscale[i] = r;
      pbuf[wave][m][ln15] = f2bf(p0);
      pbuf[wave][m][16 + ln15] = f2bf(p1);
    }
#pragma unroll
    for (int c = 0; c < 4; ++c)
#pragma unroll
      for (int i = 0; i < 8; ++i) accO[c][i] *= rscale[i];

    asm volatile("s_wait_dscnt 0" ::: "memory");

    // P as A fragment from LDS (C-layout -> A-layout conversion)
    v16bf aP;
    {
      const __bf16* Pr = &pbuf[wave][ln15][0];
#pragma unroll
      for (int j = 0; j < 8; ++j) aP[j] = Pr[hi * 16 + j];
#pragma unroll
      for (int j = 0; j < 8; ++j) aP[8 + j] = Pr[hi * 16 + 8 + j];
    }
    // O += P(16x32) @ V(32x64); V fragment contiguous thanks to vt repack
#pragma unroll
    for (int c = 0; c < 4; ++c) {
      const __bf16* Vr = vbase + (size_t)(c * 16 + ln15) * SSQ + kt + hi * 16;
      v16bf bv;
#pragma unroll
      for (int j = 0; j < 16; ++j) bv[j] = Vr[j];
      accO[c] = wmma_bf16(aP, bv, accO[c]);
    }
  }

  // finalize: divide by row sums, write bf16
#pragma unroll
  for (int c = 0; c < 4; ++c)
#pragma unroll
    for (int i = 0; i < 8; ++i) {
      int m = hi * 8 + i;
      float v = accO[c][i] / rowSum[i];
      attnOut[(bsBase + qBase + m) * EE + h * HDIM + c * 16 + ln15] = f2bf(v);
    }
}

// ---------------------------------------------------------------------------
// x = LayerNorm(x + o) * g + b   (o may be null -> plain LN of x). In place.
// One block (256 threads) per row of E=512.
// ---------------------------------------------------------------------------
__global__ void __launch_bounds__(256) add_ln_kernel(
    float* __restrict__ x, const float* __restrict__ o,
    const float* __restrict__ g, const float* __restrict__ be,
    __bf16* __restrict__ xbf) {
  int row = blockIdx.x;
  int tid = threadIdx.x;
  size_t base = (size_t)row * EE;
  float v0 = x[base + tid];
  float v1 = x[base + tid + 256];
  if (o) { v0 += o[base + tid]; v1 += o[base + tid + 256]; }
  float s = v0 + v1, ss = v0 * v0 + v1 * v1;
#pragma unroll
  for (int off = 1; off < 32; off <<= 1) {
    s += __shfl_xor(s, off, 32);
    ss += __shfl_xor(ss, off, 32);
  }
  __shared__ float rs[8], rss[8];
  __shared__ float mu_s, ri_s;
  int wv = tid >> 5, lane = tid & 31;
  if (lane == 0) { rs[wv] = s; rss[wv] = ss; }
  __syncthreads();
  if (tid == 0) {
    float ts = 0.f, tss = 0.f;
    for (int w = 0; w < 8; ++w) { ts += rs[w]; tss += rss[w]; }
    float mu = ts / (float)EE;
    float var = tss / (float)EE - mu * mu;
    mu_s = mu;
    ri_s = rsqrtf(var + 1e-5f);
  }
  __syncthreads();
  float mu = mu_s, ri = ri_s;
  float y0 = (v0 - mu) * ri * g[tid] + be[tid];
  float y1 = (v1 - mu) * ri * g[tid + 256] + be[tid + 256];
  x[base + tid] = y0;
  x[base + tid + 256] = y1;
  if (xbf) { xbf[base + tid] = f2bf(y0); xbf[base + tid + 256] = f2bf(y1); }
}

// ---------------------------------------------------------------------------
// Ring scatter: out1[b, idx-1, :] = out1raw[b, s, :] for ring-start tokens
// ---------------------------------------------------------------------------
__global__ void __launch_bounds__(256) ring_scatter_kernel(
    const int* __restrict__ seq, const float* __restrict__ out1raw,
    float* __restrict__ out1) {
  int b = blockIdx.x;
  __shared__ int idxs[SSQ];
  for (int i = threadIdx.x; i < NRING * EE; i += blockDim.x)
    out1[(size_t)b * NRING * EE + i] = 0.0f;
  if (threadIdx.x == 0) {
    int c = 0;
    for (int s = 0; s < SSQ; ++s) {
      int id = 0;
      if (seq[(size_t)b * SSQ + s] == RING_ID) {
        ++c;
        id = (c <= NRING) ? c : 0;
      }
      idxs[s] = id;
    }
  }
  __syncthreads();
  for (int s = 0; s < SSQ; ++s) {
    int id = idxs[s];
    if (id >= 1) {
      for (int e = threadIdx.x; e < EE; e += blockDim.x)
        out1[((size_t)b * NRING + (id - 1)) * EE + e] =
            out1raw[((size_t)b * SSQ + s) * EE + e];
    }
  }
}

// ---------------------------------------------------------------------------
// logits1[b,s,k] = E^-0.5 * dot(out0[b,s,:], out1[b,k,:]) -> logits[...,80+k]
// ---------------------------------------------------------------------------
__global__ void __launch_bounds__(256) logits1_kernel(
    const float* __restrict__ out0, const float* __restrict__ out1,
    float* __restrict__ logits) {
  long t = (long)blockIdx.x * blockDim.x + threadIdx.x;
  if (t >= (long)BB * SSQ * NRING) return;
  int k = (int)(t % NRING);
  long r = t / NRING;
  int s = (int)(r % SSQ);
  int b = (int)(r / SSQ);
  const float* a = out0 + ((size_t)b * SSQ + s) * EE;
  const float* c = out1 + ((size_t)b * NRING + k) * EE;
  float acc = 0.0f;
  for (int e = 0; e < EE; ++e) acc += a[e] * c[e];
  logits[((size_t)b * SSQ + s) * VV + NGEN + k] = acc * 0.04419417382415922f;
}

// ---------------------------------------------------------------------------
// Apply boolean masks: logits = NEG where graph or valence mask set
// ---------------------------------------------------------------------------
__global__ void __launch_bounds__(256) mask_apply_kernel(
    const unsigned char* __restrict__ gm, const unsigned char* __restrict__ vm,
    float* __restrict__ logits, long n) {
  long t = (long)blockIdx.x * blockDim.x + threadIdx.x;
  if (t >= n) return;
  if (gm[t] || vm[t]) logits[t] = NEGV;
}

// ---------------------------------------------------------------------------
extern "C" void kernel_launch(void* const* d_in, const int* in_sizes, int n_in,
                              void* d_out, int out_size, void* d_ws, size_t ws_size,
                              hipStream_t stream) {
  (void)in_sizes; (void)n_in; (void)out_size; (void)ws_size;
  const int* seq = (const int*)d_in[0];
  const int* cnt = (const int*)d_in[1];
  const unsigned char* gmask = (const unsigned char*)d_in[2];
  const unsigned char* vmask = (const unsigned char*)d_in[3];
  const int* lin = (const int*)d_in[4];
  const int* up = (const int*)d_in[5];
  const int* down = (const int*)d_in[6];
  const float* tok_emb = (const float*)d_in[7];
  const float* count_emb = (const float*)d_in[8];
  const float* lemb = (const float*)d_in[9];
  const float* uemb = (const float*)d_in[10];
  const float* demb = (const float*)d_in[11];
  const float* qkv_w = (const float*)d_in[12];
  const float* qkv_b = (const float*)d_in[13];
  const float* out_w = (const float*)d_in[14];
  const float* out_b = (const float*)d_in[15];
  const float* ln1_g = (const float*)d_in[16];
  const float* ln1_b = (const float*)d_in[17];
  const float* lin1_w = (const float*)d_in[18];
  const float* lin1_b = (const float*)d_in[19];
  const float* lin2_w = (const float*)d_in[20];
  const float* lin2_b = (const float*)d_in[21];
  const float* ln2_g = (const float*)d_in[22];
  const float* ln2_b = (const float*)d_in[23];
  const float* enc_g = (const float*)d_in[24];
  const float* enc_b = (const float*)d_in[25];
  const float* gen_w = (const float*)d_in[26];
  const float* gen_b = (const float*)d_in[27];
  const float* r0_w = (const float*)d_in[28];
  const float* r0_b = (const float*)d_in[29];
  const float* r1_w = (const float*)d_in[30];
  const float* r1_b = (const float*)d_in[31];
  float* logits = (float*)d_out;

  char* ws = (char*)d_ws;
  size_t off = 0;
  auto alloc = [&](size_t bytes) -> char* {
    char* p = ws + off;
    off = (off + bytes + 255) & ~(size_t)255;
    return p;
  };
  float* xf = (float*)alloc((size_t)MROWS * EE * 4);
  __bf16* xbf = (__bf16*)alloc((size_t)MROWS * EE * 2);
  float* of = (float*)alloc((size_t)MROWS * EE * 4);
  __bf16* qkvbf = (__bf16*)alloc((size_t)MROWS * 3 * EE * 2);
  __bf16* attnbf = (__bf16*)alloc((size_t)MROWS * EE * 2);
  __bf16* vtbf = (__bf16*)alloc((size_t)BB * HH * HDIM * SSQ * 2);
  __bf16* hbf = (__bf16*)alloc((size_t)MROWS * FF * 2);
  float* maskbuf = (float*)alloc((size_t)BB * HH * SSQ * SSQ * 4);
  float* out0f = (float*)alloc((size_t)MROWS * EE * 4);
  float* out1raw = (float*)alloc((size_t)MROWS * EE * 4);
  float* out1f = (float*)alloc((size_t)BB * NRING * EE * 4);
  __bf16* wqkv = (__bf16*)alloc((size_t)LLAY * 3 * EE * EE * 2);
  __bf16* wout = (__bf16*)alloc((size_t)LLAY * EE * EE * 2);
  __bf16* wl1 = (__bf16*)alloc((size_t)LLAY * FF * EE * 2);
  __bf16* wl2 = (__bf16*)alloc((size_t)LLAY * EE * FF * 2);
  __bf16* wgen = (__bf16*)alloc((size_t)NGEN * EE * 2);
  __bf16* wr0 = (__bf16*)alloc((size_t)EE * EE * 2);
  __bf16* wr1 = (__bf16*)alloc((size_t)EE * EE * 2);

  auto cvt = [&](const float* src, __bf16* dst, long n) {
    cvt_kernel<<<dim3((unsigned)((n + 255) / 256)), dim3(256), 0, stream>>>(src, dst, n);
  };
  cvt(qkv_w, wqkv, (long)LLAY * 3 * EE * EE);
  cvt(out_w, wout, (long)LLAY * EE * EE);
  cvt(lin1_w, wl1, (long)LLAY * FF * EE);
  cvt(lin2_w, wl2, (long)LLAY * EE * FF);
  cvt(gen_w, wgen, (long)NGEN * EE);
  cvt(r0_w, wr0, (long)EE * EE);
  cvt(r1_w, wr1, (long)EE * EE);

  {
    long n = (long)MROWS * EE;
    embed_kernel<<<dim3((unsigned)((n + 255) / 256)), dim3(256), 0, stream>>>(
        seq, cnt, tok_emb, count_emb, xf, xbf);
  }
  {
    long n = (long)BB * SSQ * SSQ;
    build_mask_kernel<<<dim3((unsigned)((n + 255) / 256)), dim3(256), 0, stream>>>(
        lin, up, down, lemb, uemb, demb, seq, maskbuf);
  }

  dim3 blk(128);
  for (int l = 0; l < LLAY; ++l) {
    // QKV projection: (8192,512) x (1536,512)^T -> bf16
    wmma_gemm_kernel<true><<<dim3(3 * EE / 64, MROWS / 64), blk, 0, stream>>>(
        xbf, wqkv + (size_t)l * 3 * EE * EE, qkv_b + (size_t)l * 3 * EE,
        nullptr, qkvbf, MROWS, 3 * EE, EE, 3 * EE, 0);
    // V repack for contiguous P@V fragments
    {
      long n = (long)BB * HH * HDIM * SSQ;
      repack_v_kernel<<<dim3((unsigned)((n + 255) / 256)), dim3(256), 0, stream>>>(qkvbf, vtbf);
    }
    // fused attention
    attn_kernel<<<dim3(BB * HH * (SSQ / 16) / 4), blk, 0, stream>>>(qkvbf, vtbf, maskbuf, attnbf);
    // output projection -> f32 (residual input)
    wmma_gemm_kernel<true><<<dim3(EE / 64, MROWS / 64), blk, 0, stream>>>(
        attnbf, wout + (size_t)l * EE * EE, out_b + (size_t)l * EE,
        of, nullptr, MROWS, EE, EE, EE, 0);
    add_ln_kernel<<<dim3(MROWS), dim3(256), 0, stream>>>(
        xf, of, ln1_g + (size_t)l * EE, ln1_b + (size_t)l * EE, xbf);
    // FFN up + GELU -> bf16
    wmma_gemm_kernel<true><<<dim3(FF / 64, MROWS / 64), blk, 0, stream>>>(
        xbf, wl1 + (size_t)l * FF * EE, lin1_b + (size_t)l * FF,
        nullptr, hbf, MROWS, FF, EE, FF, 1);
    // FFN down -> f32
    wmma_gemm_kernel<true><<<dim3(EE / 64, MROWS / 64), blk, 0, stream>>>(
        hbf, wl2 + (size_t)l * EE * FF, lin2_b + (size_t)l * EE,
        of, nullptr, MROWS, EE, FF, EE, 0);
    add_ln_kernel<<<dim3(MROWS), dim3(256), 0, stream>>>(
        xf, of, ln2_g + (size_t)l * EE, ln2_b + (size_t)l * EE, xbf);
  }

  // final encoder LN (no residual)
  add_ln_kernel<<<dim3(MROWS), dim3(256), 0, stream>>>(xf, nullptr, enc_g, enc_b, xbf);

  // logits0 -> d_out columns [0,80), ldc = 100 (edge tile: FULL=false)
  wmma_gemm_kernel<false><<<dim3((NGEN + 63) / 64, MROWS / 64), blk, 0, stream>>>(
      xbf, wgen, gen_b, logits, nullptr, MROWS, NGEN, EE, VV, 0);
  // ring heads
  wmma_gemm_kernel<true><<<dim3(EE / 64, MROWS / 64), blk, 0, stream>>>(
      xbf, wr0, r0_b, out0f, nullptr, MROWS, EE, EE, EE, 0);
  wmma_gemm_kernel<true><<<dim3(EE / 64, MROWS / 64), blk, 0, stream>>>(
      xbf, wr1, r1_b, out1raw, nullptr, MROWS, EE, EE, EE, 0);

  ring_scatter_kernel<<<dim3(BB), dim3(256), 0, stream>>>(seq, out1raw, out1f);
  {
    long n = (long)BB * SSQ * NRING;
    logits1_kernel<<<dim3((unsigned)((n + 255) / 256)), dim3(256), 0, stream>>>(
        out0f, out1f, logits);
  }
  {
    long n = (long)BB * SSQ * VV;
    mask_apply_kernel<<<dim3((unsigned)((n + 255) / 256)), dim3(256), 0, stream>>>(
        gmask, vmask, logits, n);
  }
}
